// Head_11519102288146
// MI455X (gfx1250) — matrix operand order repaired and checked by reading, not compiled, not run on hardware
//
#include <hip/hip_runtime.h>
#include <stdint.h>

// Problem constants (from reference): B=512, T=256, C=512, H=64
#define BB 512
#define TT 256
#define CC 512
#define HH 64

typedef __bf16        v16bf __attribute__((ext_vector_type(16)));
typedef float         v8f   __attribute__((ext_vector_type(8)));
typedef unsigned int  u32x4 __attribute__((ext_vector_type(4)));

union AV {                 // 32 bytes: one WMMA bf16 A or B operand per lane
  u32x4  u[2];
  v16bf  v;
  __bf16 h[16];
};

union PK8 {                // 16 bytes: 8 packed bf16 for a b128 store
  u32x4  u;
  __bf16 h[8];
};

static __device__ __forceinline__ v8f wmma_bf16(v16bf a, v16bf b, v8f c) {
  // D = A(16x32 bf16) * B(32x16 bf16) + C(16x16 f32)
  return __builtin_amdgcn_wmma_f32_16x16x32_bf16(false, a, false, b, (short)0, c,
                                                 false, false);
}

// ---------------------------------------------------------------------------
// Kernel 1: repack Wq/Wk/Wv (C x H fp32) into per-lane WMMA-B layout, bf16.
// pW[((mat*16 + kt)*4 + nt)*32 + lane][i] = W[kt*32 + (lane>=16?16:0) + i][nt*16 + (lane&15)]
// ---------------------------------------------------------------------------
__global__ void pack_w_kernel(const float* __restrict__ Wq,
                              const float* __restrict__ Wk,
                              const float* __restrict__ Wv,
                              __bf16* __restrict__ pW) {
  int idx = blockIdx.x * 256 + threadIdx.x;           // 3*16*4*32 = 6144 total
  if (idx >= 3 * 16 * 4 * 32) return;
  int lane = idx & 31;
  int nt   = (idx >> 5) & 3;
  int kt   = (idx >> 7) & 15;
  int mat  = idx >> 11;
  const float* W = (mat == 0) ? Wq : (mat == 1) ? Wk : Wv;
  int n     = nt * 16 + (lane & 15);
  int kbase = kt * 32 + (lane >> 4) * 16;
  __bf16* dst = pW + (size_t)idx * 16;
#pragma unroll
  for (int i = 0; i < 16; ++i)
    dst[i] = (__bf16)W[(size_t)(kbase + i) * HH + n];
}

// ---------------------------------------------------------------------------
// Kernel 2: projections. One wave = one (batch, 16-row tile): computes
// q (scaled by H^-1/2, row-major bf16), k (row-major bf16), v^T ([H][T] bf16).
// ---------------------------------------------------------------------------
__global__ __launch_bounds__(128) void proj_kernel(
    const float* __restrict__ x, const __bf16* __restrict__ pW,
    __bf16* __restrict__ qo, __bf16* __restrict__ ko,
    __bf16* __restrict__ vo) {
  const int lane = threadIdx.x & 31;
  const int wave = threadIdx.x >> 5;
  const int w    = blockIdx.x * 4 + wave;   // 0..8191
  const int b    = w >> 4;
  const int t0   = (w & 15) * 16;
  const int m    = lane & 15;
  const int hl   = lane >> 4;

  const float* xrow = x + ((size_t)b * TT + (t0 + m)) * CC;

  v8f acc[3][4];
#pragma unroll
  for (int i = 0; i < 3; ++i)
#pragma unroll
    for (int j = 0; j < 4; ++j) acc[i][j] = v8f{0.f, 0.f, 0.f, 0.f, 0.f, 0.f, 0.f, 0.f};

  for (int kt = 0; kt < 16; ++kt) {
    // A operand (16x32 of x, bf16): lane<16 holds K {0..15}, lane>=16 {16..31}
    const int kb0 = kt * 32 + hl * 16;
    const int kb1 = kb0 + 8;
    float4 f0a = *(const float4*)(xrow + kb0);
    float4 f0b = *(const float4*)(xrow + kb0 + 4);
    float4 f1a = *(const float4*)(xrow + kb1);
    float4 f1b = *(const float4*)(xrow + kb1 + 4);
    float fa[16] = {f0a.x, f0a.y, f0a.z, f0a.w, f0b.x, f0b.y, f0b.z, f0b.w,
                    f1a.x, f1a.y, f1a.z, f1a.w, f1b.x, f1b.y, f1b.z, f1b.w};
    AV a;
#pragma unroll
    for (int i = 0; i < 16; ++i) a.h[i] = (__bf16)fa[i];   // native v_cvt

#pragma unroll
    for (int mat = 0; mat < 3; ++mat) {
#pragma unroll
      for (int nt = 0; nt < 4; ++nt) {
        const u32x4* bp = (const u32x4*)(pW +
            ((((size_t)mat * 16 + kt) * 4 + nt) * 32 + lane) * 16);
        AV bb;
        bb.u[0] = bp[0];
        bb.u[1] = bp[1];
        acc[mat][nt] = wmma_bf16(a.v, bb.v, acc[mat][nt]);
      }
    }
  }

  // C-layout: element (row = r + 8*hl, col = nt*16 + m) = acc[mat][nt][r]
  const float scale = 0.125f;  // H^-1/2, folded into q
#pragma unroll
  for (int nt = 0; nt < 4; ++nt) {
#pragma unroll
    for (int r = 0; r < 8; ++r) {
      size_t off = ((size_t)b * TT + t0 + r + 8 * hl) * HH + nt * 16 + m;
      qo[off] = (__bf16)(acc[0][nt][r] * scale);
      ko[off] = (__bf16)acc[1][nt][r];
    }
  }
  // v^T: per lane, rows t0+8*hl..+7 of column h = nt*16+m are contiguous -> b128 store
#pragma unroll
  for (int nt = 0; nt < 4; ++nt) {
    PK8 pv;
#pragma unroll
    for (int j = 0; j < 8; ++j) pv.h[j] = (__bf16)acc[2][nt][j];
    *(u32x4*)(vo + ((size_t)b * HH + nt * 16 + m) * TT + t0 + 8 * hl) = pv.u;
  }
}

// ---------------------------------------------------------------------------
// Kernel 3: flash attention. One wave = one (batch, 16-query tile), iterating
// over 32-key chunks with online softmax. P goes through per-wave LDS to be
// reshaped from WMMA C-layout to A-layout (DS ops are in-order per wave).
// ---------------------------------------------------------------------------
__global__ __launch_bounds__(128) void attn_kernel(
    const __bf16* __restrict__ qm, const __bf16* __restrict__ km,
    const __bf16* __restrict__ vT, float* __restrict__ out) {
  __shared__ __bf16 Plds[4][16 * 32];  // 1KB per wave
  const int lane = threadIdx.x & 31;
  const int wave = threadIdx.x >> 5;
  const int w    = blockIdx.x * 4 + wave;
  const int b    = w >> 4;
  const int qt   = w & 15;
  const int m    = lane & 15;
  const int hl   = lane >> 4;

  // A operand for Q (16 rows x K=64): kept in registers for the whole tile
  AV aq[2];
  const __bf16* qrow = qm + ((size_t)b * TT + qt * 16 + m) * HH;
#pragma unroll
  for (int kt = 0; kt < 2; ++kt) {
    int kb = kt * 32 + hl * 16;
    aq[kt].u[0] = *(const u32x4*)(qrow + kb);
    aq[kt].u[1] = *(const u32x4*)(qrow + kb + 8);
  }

  v8f O[4];
#pragma unroll
  for (int nt = 0; nt < 4; ++nt) O[nt] = v8f{0.f, 0.f, 0.f, 0.f, 0.f, 0.f, 0.f, 0.f};
  float mrow[8], lrow[8];
#pragma unroll
  for (int r = 0; r < 8; ++r) { mrow[r] = -3.0e38f; lrow[r] = 0.f; }

  const float LOG2E   = 1.44269504f;
  const int   nChunks = qt / 2 + 1;  // ceil((qt+1)*16 / 32)
  __bf16* pl = Plds[wave];

  for (int ch = 0; ch < nChunks; ++ch) {
    const int tb = ch * 32;
    v8f S[2];
    S[0] = v8f{0.f, 0.f, 0.f, 0.f, 0.f, 0.f, 0.f, 0.f};
    S[1] = v8f{0.f, 0.f, 0.f, 0.f, 0.f, 0.f, 0.f, 0.f};
    // S = q * k^T : B operand lane = k[t' = tb+st*16+m][h-chunk contiguous]
#pragma unroll
    for (int st = 0; st < 2; ++st) {
#pragma unroll
      for (int kt = 0; kt < 2; ++kt) {
        const __bf16* kr =
            km + ((size_t)b * TT + tb + st * 16 + m) * HH + kt * 32 + hl * 16;
        AV bk;
        bk.u[0] = *(const u32x4*)kr;
        bk.u[1] = *(const u32x4*)(kr + 8);
        S[st] = wmma_bf16(aq[kt].v, bk.v, S[st]);
      }
    }
    // causal mask + row-max (row = qt*16 + r + 8*hl, col = tb + st*16 + m)
    float mx[8];
#pragma unroll
    for (int r = 0; r < 8; ++r) {
      const int tg = qt * 16 + r + 8 * hl;
      float s0 = (tb + m <= tg)      ? S[0][r] : -3.0e38f;
      float s1 = (tb + 16 + m <= tg) ? S[1][r] : -3.0e38f;
      S[0][r] = s0;
      S[1][r] = s1;
      mx[r] = fmaxf(s0, s1);
    }
#pragma unroll
    for (int off = 1; off < 16; off <<= 1)
#pragma unroll
      for (int r = 0; r < 8; ++r) mx[r] = fmaxf(mx[r], __shfl_xor(mx[r], off, 32));

    float rs[8];
#pragma unroll
    for (int r = 0; r < 8; ++r) {
      float mn    = fmaxf(mrow[r], mx[r]);
      float alpha = exp2f((mrow[r] - mn) * LOG2E);
      mrow[r]     = mn;
      float p0    = exp2f((S[0][r] - mn) * LOG2E);
      float p1    = exp2f((S[1][r] - mn) * LOG2E);
      S[0][r] = p0;
      S[1][r] = p1;
      rs[r]   = p0 + p1;
      lrow[r] *= alpha;
#pragma unroll
      for (int nt = 0; nt < 4; ++nt) O[nt][r] *= alpha;
    }
#pragma unroll
    for (int off = 1; off < 16; off <<= 1)
#pragma unroll
      for (int r = 0; r < 8; ++r) rs[r] += __shfl_xor(rs[r], off, 32);
#pragma unroll
    for (int r = 0; r < 8; ++r) lrow[r] += rs[r];

    // P (C-layout) -> LDS row-major bf16 -> reload as A-layout operand
#pragma unroll
    for (int st = 0; st < 2; ++st)
#pragma unroll
      for (int r = 0; r < 8; ++r)
        pl[(r + 8 * hl) * 32 + st * 16 + m] = (__bf16)S[st][r];
    __builtin_amdgcn_wave_barrier();
    asm volatile("" ::: "memory");
    AV ap;
    ap.u[0] = *(const u32x4*)(pl + m * 32 + hl * 16);
    ap.u[1] = *(const u32x4*)(pl + m * 32 + hl * 16 + 8);
    asm volatile("" ::: "memory");

    // O += P * v : B operand lane = vT[h = nt*16+m][t'-chunk contiguous]
#pragma unroll
    for (int nt = 0; nt < 4; ++nt) {
      const __bf16* vr =
          vT + ((size_t)b * HH + nt * 16 + m) * TT + tb + hl * 16;
      AV bv;
      bv.u[0] = *(const u32x4*)vr;
      bv.u[1] = *(const u32x4*)(vr + 8);
      O[nt] = wmma_bf16(ap.v, bv.v, O[nt]);
    }
  }

  // epilogue: out = O / l (fp32)
#pragma unroll
  for (int r = 0; r < 8; ++r) {
    float inv = 1.0f / lrow[r];
#pragma unroll
    for (int nt = 0; nt < 4; ++nt)
      out[((size_t)b * TT + qt * 16 + r + 8 * hl) * HH + nt * 16 + m] = O[nt][r] * inv;
  }
}

// ---------------------------------------------------------------------------
extern "C" void kernel_launch(void* const* d_in, const int* in_sizes, int n_in,
                              void* d_out, int out_size, void* d_ws, size_t ws_size,
                              hipStream_t stream) {
  const float* x  = (const float*)d_in[0];
  const float* Wq = (const float*)d_in[1];
  const float* Wk = (const float*)d_in[2];
  const float* Wv = (const float*)d_in[3];
  float* out      = (float*)d_out;

  char* ws = (char*)d_ws;
  // packed weights: 3*16*4*32*16 bf16 = 192 KB (rounded up to 256 KB)
  __bf16* pW      = (__bf16*)ws;
  const size_t QB = (size_t)BB * TT * HH * sizeof(__bf16);  // 16 MiB
  __bf16* qb      = (__bf16*)(ws + (1u << 18));
  __bf16* kb      = (__bf16*)(ws + (1u << 18) + QB);
  __bf16* vb      = (__bf16*)(ws + (1u << 18) + 2 * QB);

  pack_w_kernel<<<24, 256, 0, stream>>>(Wq, Wk, Wv, pW);
  proj_kernel<<<(BB * TT / 16) / 4, 128, 0, stream>>>(x, pW, qb, kb, vb);
  attn_kernel<<<(BB * TT / 16) / 4, 128, 0, stream>>>(qb, kb, vb, out);
}
